// TTTMLP_16655883174382
// MI455X (gfx1250) — compile-verified, matrix operand order checked by reading, not run on hardware
//
#include <hip/hip_runtime.h>
#include <hip/hip_bf16.h>
#include <math.h>

// ---------------------------------------------------------------------------
// TTT-MLP forward for MI455X (gfx1250), wave32 + WMMA + async-to-LDS.
//   B=4 L=2048 NH=16 HD=64 K=16 -> C=1024, F4=256, NC=128
// Pipeline:
//   K0  fp32->bf16 conversion (hs, wq, wk, wv, wo)
//   K1  bf16 WMMA GEMM (NT), double-buffered GLOBAL_LOAD_ASYNC_TO_LDS staging,
//       fused RoPE + chunked store for Q/K/V, plain store for final projection
//   K2  ttt_lr sigmoid (rank-1 eta factors)
//   K3  sequential TTT scan: 1 workgroup per (b,h), W1/W2 fp32 state in LDS
//       (needs CDNA5's 320KB/WGP), fp32 WMMA 16x16x4 recurrence, next-chunk
//       inputs prefetched via async-to-LDS overlapped with compute
//   K4  gather + post-LayerNorm -> bf16
//   K1  (mode 0) @ wo^T -> d_out fp32
// ---------------------------------------------------------------------------

typedef __attribute__((ext_vector_type(2)))  float  v2f;
typedef __attribute__((ext_vector_type(8)))  float  v8f;
typedef __attribute__((ext_vector_type(8)))  __bf16 v8bf;
typedef __attribute__((ext_vector_type(16))) __bf16 v16bf;

__device__ inline __bf16 f2bf(float f) {
  unsigned u = __builtin_bit_cast(unsigned, f);
  unsigned r = u + 0x7FFFu + ((u >> 16) & 1u);     // round-to-nearest-even
  unsigned short h = (unsigned short)(r >> 16);
  return __builtin_bit_cast(__bf16, h);
}

__device__ inline float gelu_tanh_f(float x) {
  float t = tanhf(0.79788456f * x * (1.0f + 0.044715f * x * x));
  return 0.5f * x * (1.0f + t);
}
__device__ inline float gelu_bwd_f(float x) {
  float t = tanhf(0.79788456f * x * (1.0f + 0.044715f * x * x));
  return 0.5f * x * ((1.0f - t * t) * (0.79788456f + 0.1070322243f * x * x)) + 0.5f * (1.0f + t);
}

// ----------------------- async global -> LDS helpers -----------------------
// GLOBAL_LOAD_ASYNC_TO_LDS_B128 (VGLOBAL op 98): per-lane 16B memory -> LDS,
// tracked with ASYNCcnt. LDS dest offset VGPR = low 32 bits of flat pointer.
__device__ inline unsigned lofs(const void* p) {
  return (unsigned)(unsigned long long)p;   // LDS byte offset (aperture low bits)
}
__device__ inline void async_ld_b128(unsigned lds_off, const void* gaddr) {
  unsigned long long a = (unsigned long long)gaddr;
  asm volatile("global_load_async_to_lds_b128 %0, %1, off"
               :: "v"(lds_off), "v"(a) : "memory");
}
template <int N>
__device__ inline void wait_asynccnt() {
  asm volatile("s_wait_asynccnt %0" :: "i"(N) : "memory");
}

// ------------------------- WMMA fragment helpers ---------------------------
// f32 16x16x4: A (16x4)   lane = M (0..15 both halves), VGPR k-pair per half
//              B (4x16)   lane = N, same k-pair split
//              C/D(16x16) lane = N, VGPR r -> M = r + 8*(lane>=16)
__device__ inline v8f wmma4(v2f a, v2f b, v8f c) {
  return __builtin_amdgcn_wmma_f32_16x16x4_f32(false, a, false, b, (short)0, c, false, false);
}
__device__ inline v8f wmma32bf(v16bf a, v16bf b, v8f c) {
  return __builtin_amdgcn_wmma_f32_16x16x32_bf16(false, a, false, b, (short)0, c, false, false);
}

__device__ inline v2f ldsA4(const float* A, int lda, int k0, int lane) {
  int m = lane & 15, kb = k0 + ((lane >> 4) << 1);
  v2f a; a.x = A[m * lda + kb]; a.y = A[m * lda + kb + 1]; return a;
}
__device__ inline v2f ldsB4(const float* B, int ldb, int k0, int n0, int lane) {
  int n = n0 + (lane & 15), kb = k0 + ((lane >> 4) << 1);
  v2f b; b.x = B[kb * ldb + n]; b.y = B[(kb + 1) * ldb + n]; return b;
}
__device__ inline v2f ldsB4t(const float* Bt, int ldt, int k0, int n0, int lane) {
  int n = n0 + (lane & 15), kb = k0 + ((lane >> 4) << 1);
  v2f b; b.x = Bt[n * ldt + kb]; b.y = Bt[n * ldt + kb + 1]; return b;
}
__device__ inline v2f ldsA4ct(const float* X, int ldx, int k0, int m0,
                              const float* scale, int lane) {
  int d = m0 + (lane & 15), j = k0 + ((lane >> 4) << 1);
  v2f a; a.x = scale[j] * X[j * ldx + d]; a.y = scale[j + 1] * X[(j + 1) * ldx + d]; return a;
}
__device__ inline v8f ldC16(const float* src, int ld, int n0, int lane) {
  int n = n0 + (lane & 15), rb = (lane >> 4) << 3;
  v8f c;
#pragma unroll
  for (int r = 0; r < 8; ++r) c[r] = src[(rb + r) * ld + n];
  return c;
}
__device__ inline void stD16(float* dst, int ld, int n0, v8f c, int lane) {
  int n = n0 + (lane & 15), rb = (lane >> 4) << 3;
#pragma unroll
  for (int r = 0; r < 8; ++r) dst[(rb + r) * ld + n] = c[r];
}

// bf16 16x16x32 A/B fragment from row-major bf16 [rows x 32] tile (ISA layout:
// per lane 8 contiguous at k = half*8 and 8 contiguous at k = 16 + half*8)
__device__ inline v16bf ldfrag_bf(const __bf16* base, int ld, int row0, int lane) {
  int r = row0 + (lane & 15);
  int half = lane >> 4;
  v8bf lo = *(const v8bf*)(base + r * ld + half * 8);
  v8bf hi = *(const v8bf*)(base + r * ld + 16 + half * 8);
  return __builtin_shufflevector(lo, hi, 0, 1, 2, 3, 4, 5, 6, 7,
                                 8, 9, 10, 11, 12, 13, 14, 15);
}

// --------------------------- K0: fp32 -> bf16 ------------------------------
__global__ void __launch_bounds__(256) f32_to_bf16_kernel(
    const float* __restrict__ in, __bf16* __restrict__ out, int n) {
  int i = (blockIdx.x * 256 + threadIdx.x) * 4;
  if (i < n) {
    float4 v = *(const float4*)(in + i);
    out[i + 0] = f2bf(v.x); out[i + 1] = f2bf(v.y);
    out[i + 2] = f2bf(v.z); out[i + 3] = f2bf(v.w);
  }
}

// ------------------ K1: bf16 NT GEMM  out = A @ W^T ------------------------
// A: (M x Kc) bf16 row-major, W: (N x Kc) bf16 row-major.
// Block tile 128x64, wave tile 32x32 (4 WMMA / wave / K-step).
// Double-buffered async-to-LDS staging: 3 async b128 per thread per stage.
// mode 0: out[m*N+n] fp32           (final projection)
// mode 1: chunked (B,NH,NC,K,HD) store, optional interleaved RoPE (pos = m%16)
__global__ void __launch_bounds__(256) gemm_nt_kernel(
    const __bf16* __restrict__ A, const __bf16* __restrict__ W,
    float* __restrict__ out, int M, int N, int Kc, int mode, int rope) {
  __shared__ __align__(16) __bf16 As[2][128 * 32];
  __shared__ __align__(16) __bf16 Ws[2][64 * 32];
  int tid = threadIdx.x, lane = tid & 31, wv = tid >> 5;
  int tileM = blockIdx.y * 128, tileN = blockIdx.x * 64;
  int wm = (wv >> 1) * 32;   // wave row offset inside 128x64 tile
  int wn = (wv & 1) * 32;    // wave col offset
  int arow = tid >> 1, acol = (tid & 1) * 16;  // A: 2 x b128 per thread
  int wrow = tid >> 2, wcol = (tid & 3) * 8;   // W: 1 x b128 per thread

  v8f accs[2][2];
#pragma unroll
  for (int ai = 0; ai < 2; ++ai)
#pragma unroll
    for (int bi = 0; bi < 2; ++bi)
#pragma unroll
      for (int r = 0; r < 8; ++r) accs[ai][bi][r] = 0.0f;

  // prologue: stage 0 -> buffer 0
  async_ld_b128(lofs(&As[0][arow * 32 + acol]),     &A[(size_t)(tileM + arow) * Kc + acol]);
  async_ld_b128(lofs(&As[0][arow * 32 + acol + 8]), &A[(size_t)(tileM + arow) * Kc + acol + 8]);
  async_ld_b128(lofs(&Ws[0][wrow * 32 + wcol]),     &W[(size_t)(tileN + wrow) * Kc + wcol]);

  int cur = 0;
  for (int k0 = 0; k0 < Kc; k0 += 32) {
    if (k0 + 32 < Kc) {  // issue next stage into the other buffer
      int nb = cur ^ 1, kn = k0 + 32;
      async_ld_b128(lofs(&As[nb][arow * 32 + acol]),     &A[(size_t)(tileM + arow) * Kc + kn + acol]);
      async_ld_b128(lofs(&As[nb][arow * 32 + acol + 8]), &A[(size_t)(tileM + arow) * Kc + kn + acol + 8]);
      async_ld_b128(lofs(&Ws[nb][wrow * 32 + wcol]),     &W[(size_t)(tileN + wrow) * Kc + kn + wcol]);
      wait_asynccnt<3>();   // current stage complete, next stage may fly
    } else {
      wait_asynccnt<0>();
    }
    __syncthreads();
    v16bf a0 = ldfrag_bf(As[cur], 32, wm, lane);
    v16bf a1 = ldfrag_bf(As[cur], 32, wm + 16, lane);
    v16bf b0 = ldfrag_bf(Ws[cur], 32, wn, lane);
    v16bf b1 = ldfrag_bf(Ws[cur], 32, wn + 16, lane);
    accs[0][0] = wmma32bf(a0, b0, accs[0][0]);
    accs[0][1] = wmma32bf(a0, b1, accs[0][1]);
    accs[1][0] = wmma32bf(a1, b0, accs[1][0]);
    accs[1][1] = wmma32bf(a1, b1, accs[1][1]);
    __syncthreads();
    cur ^= 1;
  }

#pragma unroll
  for (int ai = 0; ai < 2; ++ai) {
#pragma unroll
    for (int bi = 0; bi < 2; ++bi) {
      v8f acc = accs[ai][bi];
      int n = tileN + wn + bi * 16 + (lane & 15);
      int rb = (lane >> 4) << 3;
      int hd = n & 63;
      float inv_freq = 0.0f;
      if (mode == 1 && rope) inv_freq = powf(10000.0f, -(float)(hd & ~1) / 64.0f);
#pragma unroll
      for (int r = 0; r < 8; ++r) {
        int m = tileM + wm + ai * 16 + rb + r;
        float v = acc[r];
        if (mode == 0) {
          out[(size_t)m * N + n] = v;
        } else {
          if (rope) {
            float pv = __shfl_xor(v, 1, 32);          // neighbor column n^1
            float ang = (float)(m & 15) * inv_freq;   // pos = l % K
            float cs = cosf(ang), sn = sinf(ang);
            v = v * cs + ((n & 1) ? pv * sn : -pv * sn);
          }
          int b = m >> 11, l = m & 2047;
          int ch = l >> 4, kk = l & 15, h = n >> 6;
          out[((((size_t)b * 16 + h) * 128 + ch) * 16 + kk) * 64 + hd] = v;
        }
      }
    }
  }
}

// --------------------- K2: rank-1 eta column factors -----------------------
// lr[b,h,n,j] = sigmoid(<X[b, n*16+j, :], lrw[h]> + lrb[h]) * BASE_LR / HD
__global__ void __launch_bounds__(256) ttt_lr_kernel(
    const float* __restrict__ hs, const float* __restrict__ lrw,
    const float* __restrict__ lrb, float* __restrict__ lrbuf) {
  int bn = blockIdx.x;              // b*128 + chunk
  int b = bn >> 7, n = bn & 127;
  int h = threadIdx.x >> 4, j = threadIdx.x & 15;
  const float* x = hs + ((size_t)b * 2048 + n * 16 + j) * 1024;
  const float* w = lrw + (size_t)h * 1024;
  float s = 0.0f;
  for (int c = 0; c < 1024; c += 4) {
    float4 xv = *(const float4*)&x[c];
    float4 wv = *(const float4*)&w[c];
    s += xv.x * wv.x + xv.y * wv.y + xv.z * wv.z + xv.w * wv.w;
  }
  s += lrb[h];
  float sig = 1.0f / (1.0f + expf(-s));
  lrbuf[(((size_t)b * 16 + h) * 128 + n) * 16 + j] = sig * (1.0f / 64.0f);
}

// ------------------------- K3: sequential TTT scan -------------------------
// 1 workgroup (8 wave32) per (b,h). W1(64x256)+W2(256x64) fp32 state in LDS.
// Next chunk's xq/xk/xv prefetched via async-to-LDS while computing current.
__global__ void __launch_bounds__(256) ttt_scan_kernel(
    const float* __restrict__ XQ, const float* __restrict__ XK,
    const float* __restrict__ XV, const float* __restrict__ lrbuf,
    const float* __restrict__ lt, const float* __restrict__ lnW,
    const float* __restrict__ lnB, const float* __restrict__ W1_0,
    const float* __restrict__ b1_0, const float* __restrict__ W2_0,
    const float* __restrict__ b2_0, float* __restrict__ XQW) {
  __shared__ __align__(16) float W1s[64 * 256];
  __shared__ __align__(16) float W2s[256 * 64];
  __shared__ float b1s[256], b2s[64], lnw[64], lnb[64];
  __shared__ __align__(16) float xqs[2][1024], xks[2][1024], xvs[2][1024];
  __shared__ float Z1s[16 * 256], X2s[16 * 256], gZ1s[16 * 256], X2bs[16 * 256];
  __shared__ float Z2s[16 * 64], gZ2s[16 * 64], Z2bs[16 * 64];
  __shared__ float M1s[256], M2s[256];
  __shared__ float lrs[16], tokidx[16], nle[16];
  __shared__ float red1[256], red2[256];

  int tid = threadIdx.x, lane = tid & 31, wv = tid >> 5;
  int bid = blockIdx.x;             // b*16 + h
  int h = bid & 15;

  for (int i = tid; i < 64 * 256; i += 256) {
    W1s[i] = W1_0[(size_t)h * 16384 + i];
    W2s[i] = W2_0[(size_t)h * 16384 + i];
  }
  b1s[tid] = b1_0[h * 256 + tid];
  if (tid < 64) {
    b2s[tid] = b2_0[h * 64 + tid];
    lnw[tid] = lnW[h * 64 + tid];
    lnb[tid] = lnB[h * 64 + tid];
  }
  if (tid < 16) tokidx[tid] = fmaxf(1.0f / (float)(tid + 1) + lt[tid], 0.0f);

  // prologue: async-stage chunk 0 into buffer 0 (16B of each array per thread)
  size_t sbase = (size_t)bid * 128 * 1024;
  int si = tid * 4;
  async_ld_b128(lofs(&xqs[0][si]), XQ + sbase + si);
  async_ld_b128(lofs(&xks[0][si]), XK + sbase + si);
  async_ld_b128(lofs(&xvs[0][si]), XV + sbase + si);

  for (int nch = 0; nch < 128; ++nch) {
    int cur = nch & 1;
    size_t base = sbase + (size_t)nch * 1024;
    if (nch + 1 < 128) {   // overlap next chunk's fetch with this chunk's math
      int nb = cur ^ 1;
      async_ld_b128(lofs(&xqs[nb][si]), XQ + base + 1024 + si);
      async_ld_b128(lofs(&xks[nb][si]), XK + base + 1024 + si);
      async_ld_b128(lofs(&xvs[nb][si]), XV + base + 1024 + si);
    }
    if (tid < 16) {
      float lv = lrbuf[((size_t)bid * 128 + nch) * 16 + tid];
      lrs[tid] = lv;
      nle[tid] = -tokidx[15] * lv;     // -last_eta[j]
    }
    if (nch + 1 < 128) wait_asynccnt<3>();
    else               wait_asynccnt<0>();
    __syncthreads();
    const float* xq = xqs[cur];
    const float* xk = xks[cur];
    const float* xv = xvs[cur];

    // --- A: Z1 = xk @ W1 + b1 ; X2 = gelu(Z1) -----------------------------
#pragma unroll
    for (int j = 0; j < 2; ++j) {
      int n0 = wv * 32 + j * 16;
      v8f acc = {0, 0, 0, 0, 0, 0, 0, 0};
      for (int kk = 0; kk < 64; kk += 4)
        acc = wmma4(ldsA4(xk, 64, kk, lane), ldsB4(W1s, 256, kk, n0, lane), acc);
      int n = n0 + (lane & 15), rb = (lane >> 4) << 3;
#pragma unroll
      for (int r = 0; r < 8; ++r) {
        float z = acc[r] + b1s[n];
        Z1s[(rb + r) * 256 + n] = z;
        X2s[(rb + r) * 256 + n] = gelu_tanh_f(z);
      }
    }
    __syncthreads();

    // --- B: Z2 = X2 @ W2 + b2 ---------------------------------------------
    if (wv < 4) {
      int n0 = wv * 16;
      v8f acc = {0, 0, 0, 0, 0, 0, 0, 0};
      for (int kk = 0; kk < 256; kk += 4)
        acc = wmma4(ldsA4(X2s, 256, kk, lane), ldsB4(W2s, 64, kk, n0, lane), acc);
      int n = n0 + (lane & 15), rb = (lane >> 4) << 3;
#pragma unroll
      for (int r = 0; r < 8; ++r) Z2s[(rb + r) * 64 + n] = acc[r] + b2s[n];
    }
    __syncthreads();

    // --- C: gZ2 = ln_fused_l2_bwd(Z2, xv - xk, ln_w, ln_b) ----------------
    {
      int row = tid >> 4, g = tid & 15;
      float zl[4];
#pragma unroll
      for (int q = 0; q < 4; ++q) zl[q] = Z2s[row * 64 + g * 4 + q];
      red1[tid] = zl[0] + zl[1] + zl[2] + zl[3];
      __syncthreads();
      float mu = 0.0f;
#pragma unroll
      for (int q = 0; q < 16; ++q) mu += red1[(row << 4) + q];
      mu *= (1.0f / 64.0f);
      __syncthreads();
      float s = 0.0f;
#pragma unroll
      for (int q = 0; q < 4; ++q) { float d = zl[q] - mu; s += d * d; }
      red1[tid] = s;
      __syncthreads();
      float var = 0.0f;
#pragma unroll
      for (int q = 0; q < 16; ++q) var += red1[(row << 4) + q];
      float rstd = rsqrtf(var * (1.0f / 64.0f) + 1e-6f);
      __syncthreads();
      float sgo = 0.0f, sgx = 0.0f, gov[4], xhv[4];
#pragma unroll
      for (int q = 0; q < 4; ++q) {
        int c = g * 4 + q;
        float xh = (zl[q] - mu) * rstd;
        float tgt = xv[row * 64 + c] - xk[row * 64 + c];
        float go = (lnw[c] * xh + lnb[c] - tgt) * lnw[c];
        gov[q] = go; xhv[q] = xh; sgo += go; sgx += go * xh;
      }
      red1[tid] = sgo; red2[tid] = sgx;
      __syncthreads();
      float SGO = 0.0f, SGX = 0.0f;
#pragma unroll
      for (int q = 0; q < 16; ++q) { SGO += red1[(row << 4) + q]; SGX += red2[(row << 4) + q]; }
#pragma unroll
      for (int q = 0; q < 4; ++q) {
        int c = g * 4 + q;
        gZ2s[row * 64 + c] = (64.0f * gov[q] - SGO - xhv[q] * SGX) * (rstd * (1.0f / 64.0f));
      }
    }
    __syncthreads();

    // --- D: gZ1 = (gZ2 @ W2^T) * gelu_bwd(Z1) -----------------------------
#pragma unroll
    for (int j = 0; j < 2; ++j) {
      int n0 = wv * 32 + j * 16;
      v8f acc = {0, 0, 0, 0, 0, 0, 0, 0};
      for (int kk = 0; kk < 64; kk += 4)
        acc = wmma4(ldsA4(gZ2s, 64, kk, lane), ldsB4t(W2s, 64, kk, n0, lane), acc);
      int n = n0 + (lane & 15), rb = (lane >> 4) << 3;
#pragma unroll
      for (int r = 0; r < 8; ++r)
        gZ1s[(rb + r) * 256 + n] = acc[r] * gelu_bwd_f(Z1s[(rb + r) * 256 + n]);
    }
    __syncthreads();

    // --- E: M1 = -(tril(e) o (xq@xk^T + 1))  (rank-1 e folded in) ---------
    if (wv == 0) {
      v8f acc = {0, 0, 0, 0, 0, 0, 0, 0};
      for (int kk = 0; kk < 64; kk += 4)
        acc = wmma4(ldsA4(xq, 64, kk, lane), ldsB4t(xk, 64, kk, 0, lane), acc);
      int jn = lane & 15, rb = (lane >> 4) << 3;
#pragma unroll
      for (int r = 0; r < 8; ++r) {
        int i = rb + r;
        M1s[i * 16 + jn] = (i >= jn) ? -(tokidx[i] * lrs[jn] * (acc[r] + 1.0f)) : 0.0f;
      }
    }
    __syncthreads();

    // --- F: Z1_bar = xq@W1 + M1@gZ1 + b1 ; X2_bar = gelu(Z1_bar) ----------
#pragma unroll
    for (int j = 0; j < 2; ++j) {
      int n0 = wv * 32 + j * 16;
      v8f acc = {0, 0, 0, 0, 0, 0, 0, 0};
      for (int kk = 0; kk < 64; kk += 4)
        acc = wmma4(ldsA4(xq, 64, kk, lane), ldsB4(W1s, 256, kk, n0, lane), acc);
#pragma unroll
      for (int kk = 0; kk < 16; kk += 4)
        acc = wmma4(ldsA4(M1s, 16, kk, lane), ldsB4(gZ1s, 256, kk, n0, lane), acc);
      int n = n0 + (lane & 15), rb = (lane >> 4) << 3;
#pragma unroll
      for (int r = 0; r < 8; ++r)
        X2bs[(rb + r) * 256 + n] = gelu_tanh_f(acc[r] + b1s[n]);
    }
    __syncthreads();

    // --- G: M2 = -(tril(e) o (X2_bar@X2^T + 1)) ---------------------------
    if (wv == 0) {
      v8f acc = {0, 0, 0, 0, 0, 0, 0, 0};
      for (int kk = 0; kk < 256; kk += 4)
        acc = wmma4(ldsA4(X2bs, 256, kk, lane), ldsB4t(X2s, 256, kk, 0, lane), acc);
      int jn = lane & 15, rb = (lane >> 4) << 3;
#pragma unroll
      for (int r = 0; r < 8; ++r) {
        int i = rb + r;
        M2s[i * 16 + jn] = (i >= jn) ? -(tokidx[i] * lrs[jn] * (acc[r] + 1.0f)) : 0.0f;
      }
    }
    __syncthreads();

    // --- H: Z2_bar = X2_bar@W2 + M2@gZ2 + b2 ------------------------------
    if (wv < 4) {
      int n0 = wv * 16;
      v8f acc = {0, 0, 0, 0, 0, 0, 0, 0};
      for (int kk = 0; kk < 256; kk += 4)
        acc = wmma4(ldsA4(X2bs, 256, kk, lane), ldsB4(W2s, 64, kk, n0, lane), acc);
#pragma unroll
      for (int kk = 0; kk < 16; kk += 4)
        acc = wmma4(ldsA4(M2s, 16, kk, lane), ldsB4(gZ2s, 64, kk, n0, lane), acc);
      int n = n0 + (lane & 15), rb = (lane >> 4) << 3;
#pragma unroll
      for (int r = 0; r < 8; ++r) Z2bs[(rb + r) * 64 + n] = acc[r] + b2s[n];
    }
    __syncthreads();

    // --- I: out = xq + ln_fwd(Z2_bar) -------------------------------------
    {
      int row = tid >> 4, g = tid & 15;
      float zl[4];
#pragma unroll
      for (int q = 0; q < 4; ++q) zl[q] = Z2bs[row * 64 + g * 4 + q];
      red1[tid] = zl[0] + zl[1] + zl[2] + zl[3];
      __syncthreads();
      float mu = 0.0f;
#pragma unroll
      for (int q = 0; q < 16; ++q) mu += red1[(row << 4) + q];
      mu *= (1.0f / 64.0f);
      __syncthreads();
      float s = 0.0f;
#pragma unroll
      for (int q = 0; q < 4; ++q) { float d = zl[q] - mu; s += d * d; }
      red1[tid] = s;
      __syncthreads();
      float var = 0.0f;
#pragma unroll
      for (int q = 0; q < 16; ++q) var += red1[(row << 4) + q];
      float rstd = rsqrtf(var * (1.0f / 64.0f) + 1e-6f);
#pragma unroll
      for (int q = 0; q < 4; ++q) {
        int c = g * 4 + q;
        XQW[base + row * 64 + c] = xq[row * 64 + c] + lnw[c] * (zl[q] - mu) * rstd + lnb[c];
      }
    }
    __syncthreads();

    // --- J: state update ---------------------------------------------------
    // W1 -= (last_eta o xk)^T @ gZ1   (64x16 x 16x256), 64 tiles / 8 waves
#pragma unroll
    for (int p = 0; p < 8; ++p) {
      int t = wv * 8 + p, mt = t >> 4, nt = t & 15;
      float* Wtile = W1s + mt * 16 * 256;
      v8f acc = ldC16(Wtile, 256, nt * 16, lane);
#pragma unroll
      for (int kk = 0; kk < 16; kk += 4)
        acc = wmma4(ldsA4ct(xk, 64, kk, mt * 16, nle, lane),
                    ldsB4(gZ1s, 256, kk, nt * 16, lane), acc);
      stD16(Wtile, 256, nt * 16, acc, lane);
    }
    // W2 -= (last_eta o X2)^T @ gZ2   (256x16 x 16x64), 64 tiles / 8 waves
#pragma unroll
    for (int p = 0; p < 8; ++p) {
      int t = wv * 8 + p, mt = t >> 2, nt = t & 3;
      float* Wtile = W2s + mt * 16 * 64;
      v8f acc = ldC16(Wtile, 64, nt * 16, lane);
#pragma unroll
      for (int kk = 0; kk < 16; kk += 4)
        acc = wmma4(ldsA4ct(X2s, 256, kk, mt * 16, nle, lane),
                    ldsB4(gZ2s, 64, kk, nt * 16, lane), acc);
      stD16(Wtile, 64, nt * 16, acc, lane);
    }
    {
      float s = 0.0f;
#pragma unroll
      for (int j = 0; j < 16; ++j) s += nle[j] * gZ1s[j * 256 + tid];
      b1s[tid] += s;
    }
    if (tid < 64) {
      float s = 0.0f;
#pragma unroll
      for (int j = 0; j < 16; ++j) s += nle[j] * gZ2s[j * 64 + tid];
      b2s[tid] += s;
    }
    __syncthreads();
  }
}

// ------------------- K4: gather + post-LayerNorm -> bf16 -------------------
__global__ void __launch_bounds__(256) gather_postnorm_kernel(
    const float* __restrict__ XQW, const float* __restrict__ pw,
    const float* __restrict__ pb, __bf16* __restrict__ outbf) {
  __shared__ float red[256];
  int row = blockIdx.x;             // b*2048 + l
  int b = row >> 11, l = row & 2047;
  int ch = l >> 4, kk = l & 15;
  int tid = threadIdx.x;
  float local[4];
  float s = 0.0f;
#pragma unroll
  for (int q = 0; q < 4; ++q) {
    int c = tid * 4 + q, h = c >> 6, hd = c & 63;
    float v = XQW[((((size_t)b * 16 + h) * 128 + ch) * 16 + kk) * 64 + hd];
    local[q] = v; s += v;
  }
  red[tid] = s; __syncthreads();
  for (int st = 128; st > 0; st >>= 1) {
    if (tid < st) red[tid] += red[tid + st];
    __syncthreads();
  }
  float mu = red[0] * (1.0f / 1024.0f);
  __syncthreads();
  s = 0.0f;
#pragma unroll
  for (int q = 0; q < 4; ++q) { float d = local[q] - mu; s += d * d; }
  red[tid] = s; __syncthreads();
  for (int st = 128; st > 0; st >>= 1) {
    if (tid < st) red[tid] += red[tid + st];
    __syncthreads();
  }
  float rstd = rsqrtf(red[0] * (1.0f / 1024.0f) + 1e-6f);
#pragma unroll
  for (int q = 0; q < 4; ++q) {
    int c = tid * 4 + q;
    outbf[(size_t)row * 1024 + c] = f2bf(pw[c] * (local[q] - mu) * rstd + pb[c]);
  }
}

// ----------------------------- host launcher -------------------------------
extern "C" void kernel_launch(void* const* d_in, const int* in_sizes, int n_in,
                              void* d_out, int out_size, void* d_ws, size_t ws_size,
                              hipStream_t stream) {
  (void)in_sizes; (void)n_in; (void)out_size; (void)ws_size;
  const float* hs  = (const float*)d_in[0];
  const float* wq  = (const float*)d_in[1];
  const float* wk  = (const float*)d_in[2];
  const float* wvw = (const float*)d_in[3];
  const float* wo  = (const float*)d_in[4];
  const float* lt  = (const float*)d_in[5];
  const float* lrw = (const float*)d_in[6];
  const float* lrb = (const float*)d_in[7];
  const float* lnW = (const float*)d_in[8];
  const float* lnB = (const float*)d_in[9];
  const float* pw  = (const float*)d_in[10];
  const float* pb  = (const float*)d_in[11];
  const float* W1  = (const float*)d_in[12];
  const float* b1  = (const float*)d_in[13];
  const float* W2  = (const float*)d_in[14];
  const float* b2  = (const float*)d_in[15];
  float* out = (float*)d_out;

  const size_t NHS = 8388608;   // B*L*C
  const size_t NW  = 1048576;   // C*C
  char* ws = (char*)d_ws;
  size_t off = 0;
  auto carve = [&](size_t bytes) -> char* {
    char* p = ws + off;
    off += (bytes + 255) & ~(size_t)255;
    return p;
  };
  __bf16* hsb   = (__bf16*)carve(NHS * 2);
  __bf16* wqb   = (__bf16*)carve(NW * 2);
  __bf16* wkb   = (__bf16*)carve(NW * 2);
  __bf16* wvb   = (__bf16*)carve(NW * 2);
  __bf16* wob   = (__bf16*)carve(NW * 2);
  float*  XQb   = (float*)carve(NHS * 4);
  float*  XKb   = (float*)carve(NHS * 4);
  float*  XVb   = (float*)carve(NHS * 4);
  float*  XQWb  = (float*)carve(NHS * 4);
  float*  lrbuf = (float*)carve(131072 * 4);
  __bf16* nrm   = (__bf16*)carve(NHS * 2);

  f32_to_bf16_kernel<<<NHS / 1024, 256, 0, stream>>>(hs, hsb, (int)NHS);
  f32_to_bf16_kernel<<<NW / 1024, 256, 0, stream>>>(wq, wqb, (int)NW);
  f32_to_bf16_kernel<<<NW / 1024, 256, 0, stream>>>(wk, wkb, (int)NW);
  f32_to_bf16_kernel<<<NW / 1024, 256, 0, stream>>>(wvw, wvb, (int)NW);
  f32_to_bf16_kernel<<<NW / 1024, 256, 0, stream>>>(wo, wob, (int)NW);

  dim3 gg(1024 / 64, 8192 / 128);
  gemm_nt_kernel<<<gg, 256, 0, stream>>>(hsb, wqb, XQb, 8192, 1024, 1024, 1, 1);
  gemm_nt_kernel<<<gg, 256, 0, stream>>>(hsb, wkb, XKb, 8192, 1024, 1024, 1, 1);
  gemm_nt_kernel<<<gg, 256, 0, stream>>>(hsb, wvb, XVb, 8192, 1024, 1024, 1, 0);

  ttt_lr_kernel<<<512, 256, 0, stream>>>(hs, lrw, lrb, lrbuf);

  ttt_scan_kernel<<<64, 256, 0, stream>>>(XQb, XKb, XVb, lrbuf, lt, lnW, lnB,
                                          W1, b1, W2, b2, XQWb);

  gather_postnorm_kernel<<<8192, 256, 0, stream>>>(XQWb, pw, pb, nrm);

  gemm_nt_kernel<<<gg, 256, 0, stream>>>(nrm, wob, out, 8192, 1024, 1024, 0, 0);
}
